// DoubleLSTM_10101763080400
// MI455X (gfx1250) — matrix operand order repaired and checked by reading, not compiled, not run on hardware
//
#include <hip/hip_runtime.h>
#include <hip/hip_bf16.h>

// ---------------------------------------------------------------------------
// DoubleLSTM on gfx1250: bf16 WMMA fused LSTM step + TDM (tensor_load_to_lds)
// weight staging into padded LDS. B=4096, T_OBS=8, H=512, L=2, pred_len=12.
// ---------------------------------------------------------------------------

typedef __attribute__((ext_vector_type(16))) __bf16 v16bf;
typedef __attribute__((ext_vector_type(8)))  float  v8f;
typedef __attribute__((ext_vector_type(4)))  unsigned int v4u;
typedef __attribute__((ext_vector_type(8)))  int    v8i;
typedef __attribute__((ext_vector_type(4)))  int    v4i;

#define BB   4096
#define HH   512
#define KC   1024      // 2*H (concat [x,h])
#define G4H  2048      // 4*H gate rows
#define TOBS 8
#define PLEN 12
#define KB      128    // K-chunk staged in LDS
#define KB_PAD  136    // +8 bf16 (=4 DW) pad per row -> 272B pitch, bank-conflict free

#if __has_builtin(__builtin_amdgcn_tensor_load_to_lds) && __has_builtin(__builtin_amdgcn_s_wait_tensorcnt)
#define USE_TDM 1
#else
#define USE_TDM 0
#endif

__device__ __forceinline__ float sigm_f(float x) {
    return 1.0f / (1.0f + __expf(-x));
}
__device__ __forceinline__ float tanh_f(float x) {
    return 2.0f / (1.0f + __expf(-2.0f * x)) - 1.0f;
}

#if USE_TDM
// 2D TDM load: tile_d1 rows x tile_d0 bf16 elements, row stride stride_d0
// elements, into LDS at byte offset lds_off, padding 4 DWORDs after every
// 64 DWORDs stored (one tile row = 128 bf16 = 64 DW -> LDS pitch 272B).
// 6-arg toolchain variant: (v4u g0, v8i g1, v4i g2, v4i g3, v8i extra, i32 cpol).
__device__ __forceinline__ void tdm_load_2d(unsigned lds_off, const void* gaddr,
                                            unsigned tensor_d0, unsigned tensor_d1,
                                            unsigned tile_d0, unsigned tile_d1,
                                            unsigned stride_d0)
{
    unsigned long long ga = (unsigned long long)(uintptr_t)gaddr;
    v4u g0;
    g0[0] = 0x1u;                                   // count=1, user descriptor
    g0[1] = lds_off;                                // lds_addr (bytes)
    g0[2] = (unsigned)(ga & 0xffffffffu);           // global_addr[31:0]
    g0[3] = (unsigned)((ga >> 32) & 0x01ffffffu)    // global_addr[56:32]
            | 0x80000000u;                          // type=2 ("image")
    v8i g1;
    g1[0] = (int)((1u << 16)        // data_size = 2 bytes
                | (1u << 20)        // pad_enable
                | (5u << 22)        // pad_interval: 64 DWORDs
                | (3u << 25));      // pad_amount: 4 DWORDs
    g1[1] = (int)((tensor_d0 & 0xffffu) << 16);                          // dim0[15:0]
    g1[2] = (int)(((tensor_d0 >> 16) & 0xffffu) | ((tensor_d1 & 0xffffu) << 16));
    g1[3] = (int)(((tensor_d1 >> 16) & 0xffffu) | ((tile_d0 & 0xffffu) << 16));
    g1[4] = (int)(tile_d1 & 0xffffu);               // tile_dim1; tile_dim2=0
    g1[5] = (int)stride_d0;                         // tensor_dim0_stride[31:0]
    g1[6] = 0;                                      // stride hi / dim1_stride
    g1[7] = 0;
    v4i zz4 = {0, 0, 0, 0};
    v8i zz8 = {0, 0, 0, 0, 0, 0, 0, 0};
    __builtin_amdgcn_tensor_load_to_lds(g0, g1, zz4, zz4, zz8, 0);
}
#endif

// ----------------------- weight concat + bf16 convert ----------------------
__global__ __launch_bounds__(256) void wcat_kernel(
    const float* __restrict__ Wih, const float* __restrict__ Whh,
    __bf16* __restrict__ dst)
{
    int idx = blockIdx.x * 256 + threadIdx.x;       // G4H*KC threads
    int r = idx >> 10;
    int c = idx & (KC - 1);
    float v = (c < HH) ? Wih[(size_t)r * HH + c] : Whh[(size_t)r * HH + (c - HH)];
    dst[idx] = (__bf16)v;
}

// ------------------------------- embedding ---------------------------------
__global__ __launch_bounds__(256) void embed_kernel(
    const float* __restrict__ obsVel, const float* __restrict__ mean,
    const float* __restrict__ stdv,  const float* __restrict__ encW,
    const float* __restrict__ encb,  int t, __bf16* __restrict__ dstX)
{
    int idx = blockIdx.x * 256 + threadIdx.x;       // BB*HH threads
    int b = idx >> 9;
    int h = idx & (HH - 1);
    float i0 = (obsVel[(size_t)b * TOBS * 2 + t * 2 + 0] - mean[0]) / stdv[0];
    float i1 = (obsVel[(size_t)b * TOBS * 2 + t * 2 + 1] - mean[1]) / stdv[1];
    float v = encb[h] + i0 * encW[h] + i1 * encW[HH + h];
    dstX[(size_t)b * KC + h] = (__bf16)v;
}

// --------------------------- fused LSTM step (WMMA) ------------------------
// Wave: 32 rows x 16 cols x 4 gates (8 accumulators). Block: 8 waves =
// 128 rows x 32 cols. Weights staged per 128-K chunk into LDS via TDM.
__global__ __launch_bounds__(256) void lstm_step_kernel(
    const __bf16* __restrict__ xh,    // [B][KC] read phase
    const __bf16* __restrict__ Wcat,  // [4H][KC] row-major
    const float*  __restrict__ bias,  // [4H]
    float*        __restrict__ c,     // [B][H]
    __bf16*       __restrict__ dst_h, // next-phase xh of same layer: [b][H+n]
    __bf16*       __restrict__ dst_x) // consumer xh (or null): [b][n]
{
    __shared__ __bf16 ldsB[4 * 32 * KB_PAD];        // 4 gates x 32 cols x 136

    const int tid  = threadIdx.x;
    const int wave = tid >> 5;
    const int lane = tid & 31;
    const int half = lane >> 4;
    const int lq   = lane & 15;
    const int wm   = wave & 3;                      // 4 M-groups of 32 rows
    const int wn   = wave >> 2;                     // 2 N-groups of 16 cols
    const int row0 = blockIdx.x * 128 + wm * 32;    // gridDim.x = 32
    const int col0 = blockIdx.y * 32 + wn * 16;     // gridDim.y = 16
    const int colBase = blockIdx.y * 32;

    // A fragments: rows row0+lq and row0+16+lq, K window [k+8h, k+8h+8)U[k+16+8h,...)
    const __bf16* aP0 = xh + (size_t)(row0 + lq) * KC + half * 8;
    const __bf16* aP1 = aP0 + (size_t)16 * KC;
    // B fragments from LDS: row (gate*32 + wn*16+lq), 8 bf16 at k2+half*8 and 8 at k2+16+half*8
    const __bf16* lB = ldsB + (size_t)(wn * 16 + lq) * KB_PAD + half * 8;

    v8f acc[2][4];
#pragma unroll
    for (int mt = 0; mt < 2; ++mt)
#pragma unroll
        for (int g = 0; g < 4; ++g) acc[mt][g] = (v8f){};

    const uint4 zero4 = {0u, 0u, 0u, 0u};

    for (int kc = 0; kc < KC; kc += KB) {
        // ---- stage 4 gate weight tiles (32 rows x 128 bf16) into LDS ----
        if (wave < 4) {
#if USE_TDM
            const __bf16* src = Wcat + ((size_t)(wave * HH + colBase) * KC + kc);
            tdm_load_2d((unsigned)(wave * 32 * KB_PAD * sizeof(__bf16)), src,
                        /*tensor_d0=*/KC, /*tensor_d1=*/G4H,
                        /*tile_d0=*/KB, /*tile_d1=*/32, /*stride_d0=*/KC);
            __builtin_amdgcn_s_wait_tensorcnt(0);
#else
            // fallback: cooperative copy (keeps ds path alive)
            const __bf16* src = Wcat + ((size_t)(wave * HH + colBase) * KC + kc);
            for (int i = lane; i < 32 * (KB / 8); i += 32) {
                int r = i >> 4, c8 = i & 15;        // KB/8 = 16 uint4 per row
                *(uint4*)(ldsB + (size_t)(wave * 32 + r) * KB_PAD + c8 * 8) =
                    *(const uint4*)(src + (size_t)r * KC + c8 * 8);
            }
#endif
        } else {
            // Waves 4-7: zero the 16B pad slot of each row of gate (wave-4).
            // TDM skips (doesn't write) pad bytes, so this cannot race it.
            // These dynamic-index ds_stores also make ldsB visibly written,
            // preventing LLVM from folding the fragment ds_loads to undef.
            int g = wave - 4;
            *(uint4*)(ldsB + (size_t)(g * 32 + lane) * KB_PAD + KB) = zero4;
        }
        __syncthreads();
        // Escape ldsB's address into a may-write asm: the TDM builtin only
        // receives an integer LDS offset, so without this escape LLVM proves
        // the array is never stored through and folds all ds_loads to undef.
        asm volatile("" :: "v"((const __bf16*)ldsB) : "memory");

#pragma unroll 2
        for (int k2 = 0; k2 < KB; k2 += 32) {
            v16bf a0, a1, b0, b1, b2, b3;
            ((uint4*)&a0)[0] = *(const uint4*)(aP0);
            ((uint4*)&a0)[1] = *(const uint4*)(aP0 + 16);
            ((uint4*)&a1)[0] = *(const uint4*)(aP1);
            ((uint4*)&a1)[1] = *(const uint4*)(aP1 + 16);
            const __bf16* lb = lB + k2;
            ((uint4*)&b0)[0] = *(const uint4*)(lb + 0 * 32 * KB_PAD);
            ((uint4*)&b0)[1] = *(const uint4*)(lb + 0 * 32 * KB_PAD + 16);
            ((uint4*)&b1)[0] = *(const uint4*)(lb + 1 * 32 * KB_PAD);
            ((uint4*)&b1)[1] = *(const uint4*)(lb + 1 * 32 * KB_PAD + 16);
            ((uint4*)&b2)[0] = *(const uint4*)(lb + 2 * 32 * KB_PAD);
            ((uint4*)&b2)[1] = *(const uint4*)(lb + 2 * 32 * KB_PAD + 16);
            ((uint4*)&b3)[0] = *(const uint4*)(lb + 3 * 32 * KB_PAD);
            ((uint4*)&b3)[1] = *(const uint4*)(lb + 3 * 32 * KB_PAD + 16);

            acc[0][0] = __builtin_amdgcn_wmma_f32_16x16x32_bf16(false, a0, false, b0,
                                                                (short)0, acc[0][0], false, false);
            acc[0][1] = __builtin_amdgcn_wmma_f32_16x16x32_bf16(false, a0, false, b1,
                                                                (short)0, acc[0][1], false, false);
            acc[0][2] = __builtin_amdgcn_wmma_f32_16x16x32_bf16(false, a0, false, b2,
                                                                (short)0, acc[0][2], false, false);
            acc[0][3] = __builtin_amdgcn_wmma_f32_16x16x32_bf16(false, a0, false, b3,
                                                                (short)0, acc[0][3], false, false);
            acc[1][0] = __builtin_amdgcn_wmma_f32_16x16x32_bf16(false, a1, false, b0,
                                                                (short)0, acc[1][0], false, false);
            acc[1][1] = __builtin_amdgcn_wmma_f32_16x16x32_bf16(false, a1, false, b1,
                                                                (short)0, acc[1][1], false, false);
            acc[1][2] = __builtin_amdgcn_wmma_f32_16x16x32_bf16(false, a1, false, b2,
                                                                (short)0, acc[1][2], false, false);
            acc[1][3] = __builtin_amdgcn_wmma_f32_16x16x32_bf16(false, a1, false, b3,
                                                                (short)0, acc[1][3], false, false);
            aP0 += 32; aP1 += 32;
        }
        asm volatile("" :: "v"((const __bf16*)ldsB) : "memory");
        __syncthreads();
    }

    // ---- gate math in-register: VGPR r -> M = mt*16 + half*8 + r, N = lq ----
    const int n  = col0 + lq;
    const float bi = bias[0 * HH + n];
    const float bf = bias[1 * HH + n];
    const float bg = bias[2 * HH + n];
    const float bo = bias[3 * HH + n];

#pragma unroll
    for (int mt = 0; mt < 2; ++mt) {
#pragma unroll
        for (int r = 0; r < 8; ++r) {
            const int row = row0 + mt * 16 + half * 8 + r;
            float gi = acc[mt][0][r] + bi;
            float gf = acc[mt][1][r] + bf;
            float gg = acc[mt][2][r] + bg;
            float go = acc[mt][3][r] + bo;
            size_t cidx = (size_t)row * HH + n;
            float cn = sigm_f(gf) * c[cidx] + sigm_f(gi) * tanh_f(gg);
            float hn = sigm_f(go) * tanh_f(cn);
            c[cidx] = cn;
            __bf16 hb = (__bf16)hn;
            dst_h[(size_t)row * KC + HH + n] = hb;
            if (dst_x) dst_x[(size_t)row * KC + n] = hb;
        }
    }
}

// ------------------------- decode projection (H -> 2) ----------------------
__global__ __launch_bounds__(256) void outvel_kernel(
    const __bf16* __restrict__ hsrc, const float* __restrict__ decW,
    const float* __restrict__ decb,  float* __restrict__ outVel, int t)
{
    int idx = blockIdx.x * 256 + threadIdx.x;   // BB*2 threads
    int b = idx >> 1;
    int d = idx & 1;
    float s = decb[d];
    const __bf16* hp = hsrc + (size_t)b * KC;
#pragma unroll 8
    for (int k = 0; k < HH; ++k) s += (float)hp[k] * decW[k * 2 + d];
    outVel[(size_t)b * (PLEN * 2) + t * 2 + d] = s;
}

// ----------------------- scale + cumsum + last-obs add ---------------------
__global__ __launch_bounds__(256) void final_kernel(
    const float* __restrict__ outVel, const float* __restrict__ stdv,
    const float* __restrict__ mean,   const float* __restrict__ obs,
    float* __restrict__ out)
{
    int b = blockIdx.x * 256 + threadIdx.x;     // BB threads
    float a0 = obs[(size_t)b * TOBS * 2 + (TOBS - 1) * 2 + 0];
    float a1 = obs[(size_t)b * TOBS * 2 + (TOBS - 1) * 2 + 1];
    const float s0 = stdv[0], s1 = stdv[1], m0 = mean[0], m1 = mean[1];
#pragma unroll
    for (int t = 0; t < PLEN; ++t) {
        a0 += outVel[(size_t)b * (PLEN * 2) + t * 2 + 0] * s0 + m0;
        a1 += outVel[(size_t)b * (PLEN * 2) + t * 2 + 1] * s1 + m1;
        out[(size_t)b * (PLEN * 2) + t * 2 + 0] = a0;
        out[(size_t)b * (PLEN * 2) + t * 2 + 1] = a1;
    }
}

// ---------------------------------------------------------------------------
extern "C" void kernel_launch(void* const* d_in, const int* in_sizes, int n_in,
                              void* d_out, int out_size, void* d_ws, size_t ws_size,
                              hipStream_t stream) {
    (void)in_sizes; (void)n_in; (void)out_size; (void)ws_size;
    const float* obs      = (const float*)d_in[0];
    const float* obsVel   = (const float*)d_in[1];
    const float* mean     = (const float*)d_in[2];
    const float* stdv     = (const float*)d_in[3];
    /* d_in[4] = pred_len (==12, hardcoded) */
    const float* encW     = (const float*)d_in[5];
    const float* encb     = (const float*)d_in[6];
    const float* decW     = (const float*)d_in[7];
    const float* decb     = (const float*)d_in[8];
    const float* lstm_Wih = (const float*)d_in[9];
    const float* lstm_Whh = (const float*)d_in[10];
    const float* lstm_b   = (const float*)d_in[11];
    const float* cell_Wih = (const float*)d_in[12];
    const float* cell_Whh = (const float*)d_in[13];
    const float* cell_b   = (const float*)d_in[14];
    float* out = (float*)d_out;

    char* ws = (char*)d_ws;
    size_t off = 0;
    auto carve = [&](size_t bytes) -> void* {
        void* p = ws + off;
        off += (bytes + 255) & ~(size_t)255;
        return p;
    };
    const size_t WCAT_B = (size_t)G4H * KC * sizeof(__bf16);   // 4 MB
    const size_t XH_B   = (size_t)BB * KC * sizeof(__bf16);    // 8 MB
    const size_t C_B    = (size_t)BB * HH * sizeof(float);     // 8 MB

    __bf16* WL[2]  = { (__bf16*)carve(WCAT_B), (__bf16*)carve(WCAT_B) };
    __bf16* WC[2]  = { (__bf16*)carve(WCAT_B), (__bf16*)carve(WCAT_B) };
    __bf16* xh0[2] = { (__bf16*)carve(XH_B), (__bf16*)carve(XH_B) };
    __bf16* xh1[2] = { (__bf16*)carve(XH_B), (__bf16*)carve(XH_B) };
    float*  c0 = (float*)carve(C_B);
    float*  c1 = (float*)carve(C_B);
    float*  outVel = (float*)carve((size_t)BB * PLEN * 2 * sizeof(float));

    (void)hipMemsetAsync(xh0[0], 0, XH_B, stream);
    (void)hipMemsetAsync(xh0[1], 0, XH_B, stream);
    (void)hipMemsetAsync(xh1[0], 0, XH_B, stream);
    (void)hipMemsetAsync(xh1[1], 0, XH_B, stream);
    (void)hipMemsetAsync(c0, 0, C_B, stream);
    (void)hipMemsetAsync(c1, 0, C_B, stream);

    const int wcatBlocks = (G4H * KC) / 256;
    for (int l = 0; l < 2; ++l) {
        wcat_kernel<<<wcatBlocks, 256, 0, stream>>>(
            lstm_Wih + (size_t)l * G4H * HH, lstm_Whh + (size_t)l * G4H * HH, WL[l]);
        wcat_kernel<<<wcatBlocks, 256, 0, stream>>>(
            cell_Wih + (size_t)l * G4H * HH, cell_Whh + (size_t)l * G4H * HH, WC[l]);
    }

    const dim3 gGrid(BB / 128, HH / 32);     // 32 x 16 blocks
    const dim3 gBlk(256);
    const int embedBlocks = (BB * HH) / 256;

    // ---- encoder: 8 timesteps, 2 layers ----
    for (int t = 0; t < TOBS; ++t) {
        int p = t & 1, q = 1 - p;
        embed_kernel<<<embedBlocks, 256, 0, stream>>>(obsVel, mean, stdv, encW, encb,
                                                      t, xh0[p]);
        lstm_step_kernel<<<gGrid, gBlk, 0, stream>>>(
            xh0[p], WL[0], lstm_b, c0, xh0[q], xh1[p]);
        lstm_step_kernel<<<gGrid, gBlk, 0, stream>>>(
            xh1[p], WL[1], lstm_b + G4H, c1, xh1[q],
            (t == TOBS - 1) ? xh0[q] : (__bf16*)nullptr);
    }

    outvel_kernel<<<(BB * 2) / 256, 256, 0, stream>>>(xh0[0], decW, decb, outVel, 0);

    // ---- decoder: 11 steps, 2 cell layers ----
    for (int s = 1; s < PLEN; ++s) {
        int g = (TOBS - 1) + s;
        int p = g & 1, q = 1 - p;
        lstm_step_kernel<<<gGrid, gBlk, 0, stream>>>(
            xh0[p], WC[0], cell_b, c0, xh0[q], xh1[p]);
        lstm_step_kernel<<<gGrid, gBlk, 0, stream>>>(
            xh1[p], WC[1], cell_b + G4H, c1, xh1[q], xh0[q]);
        outvel_kernel<<<(BB * 2) / 256, 256, 0, stream>>>(xh0[q], decW, decb, outVel, s);
    }

    final_kernel<<<BB / 256, 256, 0, stream>>>(outVel, stdv, mean, obs, out);
}
